// SpatioTemporalGraphAttentionLayer_43748536877443
// MI455X (gfx1250) — matrix operand order repaired and checked by reading, not compiled, not run on hardware
//
#include <hip/hip_runtime.h>
#include <hip/hip_bf16.h>
#include <stdint.h>

typedef __attribute__((ext_vector_type(16))) __bf16 v16bf;
typedef __attribute__((ext_vector_type(8)))  float  v8f;
typedef __attribute__((ext_vector_type(4)))  unsigned int u32x4;

#define B_    4
#define N_    2048
#define T_    12
#define FIN   32
#define D_    384          /* T*F_IN == T*F_OUT */
#define NF    (N_ * FIN)   /* 65536 */
#define ALPHA_  0.2f
#define NEGINF_ (-9000000000000000.0f)

__device__ __forceinline__ unsigned short f32_to_bf16(float f) {
  unsigned int u = __float_as_uint(f);
  u += 0x7fffu + ((u >> 16) & 1u);           // round-to-nearest-even
  return (unsigned short)(u >> 16);
}
__device__ __forceinline__ float bf16_to_f32(unsigned short h) {
  return __uint_as_float(((unsigned int)h) << 16);
}

// ---------------------------------------------------------------------------
// K1: temporal conv along flattened N*F_IN axis, write bf16 xt in (B*N, 384)
// layout: xt[(b*N+n)*384 + t_out*32 + f]
// ---------------------------------------------------------------------------
__global__ void __launch_bounds__(256) k_conv_pack(
    const float* __restrict__ x, const float* __restrict__ cw,
    const float* __restrict__ cb, unsigned short* __restrict__ xt) {
  int idx = blockIdx.x * 256 + threadIdx.x;        // < B*T*NF
  int h  = idx & (NF - 1);
  int bt = idx >> 16;                              // idx / NF
  int to = bt % T_;
  int b  = bt / T_;
  float acc = cb[to];
#pragma unroll
  for (int k = 0; k < 3; ++k) {
    int hh = h + k - 1;
    if (hh < 0 || hh >= NF) continue;
    int n = hh >> 5, f = hh & 31;
    const float* xp = x + ((size_t)(b * N_ + n)) * (T_ * FIN) + f;
#pragma unroll
    for (int ti = 0; ti < T_; ++ti)
      acc = fmaf(cw[(to * T_ + ti) * 3 + k], xp[ti * FIN], acc);
  }
  int n = h >> 5, f = h & 31;
  xt[((size_t)(b * N_ + n)) * D_ + to * FIN + f] = f32_to_bf16(acc);
}

// ---------------------------------------------------------------------------
// K2: W (384x384 f32, row-major k,n) -> WT bf16 (n,k) for WMMA B operand
// ---------------------------------------------------------------------------
__global__ void __launch_bounds__(256) k_transpose_w(
    const float* __restrict__ W, unsigned short* __restrict__ WT) {
  int idx = blockIdx.x * 256 + threadIdx.x;        // < 384*384
  int k = idx % D_, n = idx / D_;
  WT[(size_t)n * D_ + k] = f32_to_bf16(W[(size_t)k * D_ + n]);
}

// ---------------------------------------------------------------------------
// K3: Wh = xt @ W.  64-row tile per block; each wave owns 3 feature blocks
// and 4 row-blocks (12 C tiles).  Per K step: 3 B tiles + all 4 A tiles are
// loaded into distinct registers BEFORE the 12-WMMA block, so loads clause
// together, waits overlap, and no WMMA<->load WAR hazards remain.
// Writes Wh bf16 (row, col) and WhT bf16 (b, col, node) for the attn GEMM.
// ---------------------------------------------------------------------------
__global__ void __launch_bounds__(256) k_gemm_wh(
    const unsigned short* __restrict__ xt, const unsigned short* __restrict__ WT,
    unsigned short* __restrict__ Whb, unsigned short* __restrict__ WhT) {
  int row0 = blockIdx.x * 64;                      // < 8192, 64-row tile
  int wave = threadIdx.x >> 5, lane = threadIdx.x & 31;
  int mrow = lane & 15, half = lane >> 4;
  v8f acc[4][3] = {};
  for (int k0 = 0; k0 < D_; k0 += 32) {
    // B: 32x16 bf16 per feature block; lane half holds K {0..7,16..23} / {8..15,24..31}
    v16bf Bm[3];
#pragma unroll
    for (int fb = 0; fb < 3; ++fb) {
      int n0 = wave * 48 + fb * 16;
      const unsigned short* wp = WT + (size_t)(n0 + mrow) * D_ + k0 + half * 8;
      *(u32x4*)&Bm[fb]       = *(const u32x4*)wp;
      *((u32x4*)&Bm[fb] + 1) = *(const u32x4*)(wp + 16);
    }
    // A: 16x32 bf16 x4; lane half holds K {0..7,16..23} / {8..15,24..31}
    v16bf A[4];
#pragma unroll
    for (int ib = 0; ib < 4; ++ib) {
      const unsigned short* ap =
          xt + (size_t)(row0 + ib * 16 + mrow) * D_ + k0 + half * 8;
      *(u32x4*)&A[ib]       = *(const u32x4*)ap;
      *((u32x4*)&A[ib] + 1) = *(const u32x4*)(ap + 16);
    }
#pragma unroll
    for (int ib = 0; ib < 4; ++ib)
#pragma unroll
      for (int fb = 0; fb < 3; ++fb)
        acc[ib][fb] = __builtin_amdgcn_wmma_f32_16x16x32_bf16(
            false, A[ib], false, Bm[fb], (short)0, acc[ib][fb], false, false);
  }
  int b = row0 >> 11;
#pragma unroll
  for (int ib = 0; ib < 4; ++ib) {
#pragma unroll
    for (int fb = 0; fb < 3; ++fb) {
      int col = wave * 48 + fb * 16 + mrow;
#pragma unroll
      for (int r = 0; r < 8; ++r) {
        int row = row0 + ib * 16 + half * 8 + r;   // C layout: M = half*8 + r
        unsigned short v = f32_to_bf16(acc[ib][fb][r]);
        Whb[(size_t)row * D_ + col] = v;
        WhT[((size_t)b * D_ + col) * N_ + (row & (N_ - 1))] = v;
      }
    }
  }
}

// ---------------------------------------------------------------------------
// K4: f1 = Wh.a1, f2 = Wh.a2 per row (f32 accumulate from bf16 Wh)
// ---------------------------------------------------------------------------
__global__ void __launch_bounds__(256) k_f12(
    const unsigned short* __restrict__ Whb, const float* __restrict__ a,
    float* __restrict__ f1, float* __restrict__ f2) {
  int row = blockIdx.x * 256 + threadIdx.x;        // < 8192
  const unsigned short* wp = Whb + (size_t)row * D_;
  float s1 = 0.f, s2 = 0.f;
  for (int k = 0; k < D_; ++k) {
    float w = bf16_to_f32(wp[k]);
    s1 = fmaf(w, a[k], s1);
    s2 = fmaf(w, a[D_ + k], s2);
  }
  f1[row] = s1;
  f2[row] = s2;
}

// ---------------------------------------------------------------------------
// K5: per-row softmax stats (max, 1/sum) via online logsumexp; one wave/row.
// Scores are rank-1 + mask, so the 4x2048x2048 tensor is never materialized.
// ---------------------------------------------------------------------------
__global__ void __launch_bounds__(256) k_softmax_stats(
    const float* __restrict__ f1, const float* __restrict__ f2,
    const int* __restrict__ adj, float* __restrict__ rmax,
    float* __restrict__ rinv) {
  int wave = threadIdx.x >> 5, lane = threadIdx.x & 31;
  int row = blockIdx.x * 8 + wave;                 // < 8192
  int b = row >> 11, i = row & (N_ - 1);
  float f1v = f1[row];
  float m = -__builtin_inff(), s = 0.f;
  for (int j = lane; j < N_; j += 32) {
    float e = f1v + f2[b * N_ + j];
    e = e > 0.f ? e : ALPHA_ * e;
    e = adj[(size_t)i * N_ + j] > 0 ? e : NEGINF_;
    float nm = fmaxf(m, e);
    s = s * __expf(m - nm) + __expf(e - nm);
    m = nm;
  }
#pragma unroll
  for (int off = 16; off > 0; off >>= 1) {
    float om = __shfl_xor(m, off, 32);
    float os = __shfl_xor(s, off, 32);
    float nm = fmaxf(m, om);
    s = s * __expf(m - nm) + os * __expf(om - nm);
    m = nm;
  }
  if (lane == 0) { rmax[row] = m; rinv[row] = 1.f / s; }
}

// ---------------------------------------------------------------------------
// K6: h' = softmax(e) @ Wh, fused with ELU.  64-row i-tile per block.
// p tile (64x32 bf16) built in LDS each j step (each thread: 8 contiguous
// probabilities); 8 waves each hold 12 C tiles; 3 B tiles (from WhT,
// prefetched) and 4 A tiles (ds_load_b128 x8) loaded before the WMMA block.
// ---------------------------------------------------------------------------
__global__ void __launch_bounds__(256) k_attn(
    const float* __restrict__ f1, const float* __restrict__ f2,
    const int* __restrict__ adj, const float* __restrict__ rmax,
    const float* __restrict__ rinv, const unsigned short* __restrict__ WhT,
    float* __restrict__ out) {
  __shared__ __align__(16) unsigned short pbuf[64 * 32];
  int tile = blockIdx.x;                           // < 128
  int b = tile >> 5;
  int i0 = (tile & 31) * 64;
  int row0 = b * N_ + i0;
  int tid = threadIdx.x, wave = tid >> 5, lane = tid & 31;
  int mrow = lane & 15, half = lane >> 4;
  int pi = tid >> 2, pg = (tid & 3) * 8;           // row 0..63, col group of 8
  float f1v = f1[row0 + pi];
  float mv  = rmax[row0 + pi];
  const int* adjrow = adj + (size_t)(i0 + pi) * N_;
  const float* f2b = f2 + b * N_;
  v8f acc[4][3] = {};
  for (int j0 = 0; j0 < N_; j0 += 32) {
#pragma unroll
    for (int u = 0; u < 8; ++u) {
      int jj = pg + u;
      int j = j0 + jj;
      float e = f1v + f2b[j];
      e = e > 0.f ? e : ALPHA_ * e;
      e = adjrow[j] > 0 ? e : NEGINF_;
      pbuf[pi * 32 + jj] = f32_to_bf16(__expf(e - mv));
    }
    __syncthreads();
    v16bf Bm[3];
#pragma unroll
    for (int fb = 0; fb < 3; ++fb) {
      int n0 = wave * 48 + fb * 16;
      const unsigned short* bp =
          WhT + ((size_t)b * D_ + n0 + mrow) * N_ + j0 + half * 8;
      *(u32x4*)&Bm[fb]       = *(const u32x4*)bp;
      *((u32x4*)&Bm[fb] + 1) = *(const u32x4*)(bp + 16);
      __builtin_prefetch((const void*)(bp + 32), 0, 1);       // next j-tile
    }
    v16bf A[4];
#pragma unroll
    for (int ib = 0; ib < 4; ++ib) {
      const unsigned short* ap =
          pbuf + (ib * 16 + mrow) * 32 + half * 8;            // ds_load_b128 x2
      *(u32x4*)&A[ib]       = *(const u32x4*)ap;
      *((u32x4*)&A[ib] + 1) = *(const u32x4*)(ap + 16);
    }
#pragma unroll
    for (int ib = 0; ib < 4; ++ib)
#pragma unroll
      for (int fb = 0; fb < 3; ++fb)
        acc[ib][fb] = __builtin_amdgcn_wmma_f32_16x16x32_bf16(
            false, A[ib], false, Bm[fb], (short)0, acc[ib][fb], false, false);
    __syncthreads();
  }
#pragma unroll
  for (int ib = 0; ib < 4; ++ib) {
#pragma unroll
    for (int fb = 0; fb < 3; ++fb) {
      int col = wave * 48 + fb * 16 + mrow;
#pragma unroll
      for (int r = 0; r < 8; ++r) {
        int i = i0 + ib * 16 + half * 8 + r;
        float v = acc[ib][fb][r] * rinv[b * N_ + i];
        v = v > 0.f ? v : (__expf(v) - 1.f);                  // ELU
        out[((size_t)(b * N_ + i)) * D_ + col] = v;
      }
    }
  }
}

// ---------------------------------------------------------------------------
// launch
// ---------------------------------------------------------------------------
extern "C" void kernel_launch(void* const* d_in, const int* in_sizes, int n_in,
                              void* d_out, int out_size, void* d_ws, size_t ws_size,
                              hipStream_t stream) {
  const float* x   = (const float*)d_in[0];
  const int*   adj = (const int*)d_in[1];
  const float* W   = (const float*)d_in[2];
  const float* a   = (const float*)d_in[3];
  const float* cw  = (const float*)d_in[4];
  const float* cb  = (const float*)d_in[5];
  float* out = (float*)d_out;

  char* ws = (char*)d_ws;
  unsigned short* xt   = (unsigned short*)(ws);             // 6,291,456 B
  unsigned short* WT   = (unsigned short*)(ws +  6291456);  //   294,912 B
  unsigned short* Whb  = (unsigned short*)(ws +  6586368);  // 6,291,456 B
  unsigned short* WhT  = (unsigned short*)(ws + 12877824);  // 6,291,456 B
  float* f1   = (float*)(ws + 19169280);
  float* f2   = (float*)(ws + 19202048);
  float* rmax = (float*)(ws + 19234816);
  float* rinv = (float*)(ws + 19267584);                    // end ~19.3 MB

  k_conv_pack    <<<(B_ * T_ * NF) / 256, 256, 0, stream>>>(x, cw, cb, xt);
  k_transpose_w  <<<(D_ * D_) / 256,      256, 0, stream>>>(W, WT);
  k_gemm_wh      <<<(B_ * N_) / 64,       256, 0, stream>>>(xt, WT, Whb, WhT);
  k_f12          <<<(B_ * N_) / 256,      256, 0, stream>>>(Whb, a, f1, f2);
  k_softmax_stats<<<(B_ * N_) / 8,        256, 0, stream>>>(f1, f2, adj, rmax, rinv);
  k_attn         <<<(B_ * N_) / 64,       256, 0, stream>>>(f1, f2, adj, rmax, rinv, WhT, out);
}
